// SpatioTemporalBlock_58428735095525
// MI455X (gfx1250) — matrix operand order repaired and checked by reading, not compiled, not run on hardware
//
#include <hip/hip_runtime.h>
#include <hip/hip_bf16.h>
#include <math.h>

#define N_NODES 4096
#define E_EDGES 131072
#define C_IN    128
#define HID     128
#define OUT_CH  32
#define HEADS   8
#define DH      16
#define FF      2048
#define EPS     1e-5f

typedef __attribute__((ext_vector_type(4)))  __bf16 v4bf;
typedef __attribute__((ext_vector_type(8)))  __bf16 v8bf;
typedef __attribute__((ext_vector_type(16))) __bf16 v16bf;
typedef __attribute__((ext_vector_type(8)))  float  v8f;
typedef __attribute__((ext_vector_type(4)))  unsigned int v4u;
typedef __attribute__((ext_vector_type(8)))  int v8i;
typedef __attribute__((ext_vector_type(4)))  int v4i;

// ---------------------------------------------------------------- WMMA fragment helpers
__device__ inline v16bf bf_combine(v8bf lo, v8bf hi) {
    v16bf r;
#pragma unroll
    for (int i = 0; i < 8; ++i) { r[i] = lo[i]; r[i + 8] = hi[i]; }
    return r;
}

// A fragment: 16x32 bf16 from row-major [.,128] tile.
// lane<16: row rb+lane, K = k0..k0+7 , k0+16..k0+23
// lane>=16: row rb+lane-16, K = k0+8..k0+15 , k0+24..k0+31
__device__ inline v16bf load_a_frag(const __bf16* tile, int rb, int k0, int lane) {
    int r  = rb + (lane & 15);
    int ka = k0 + ((lane >> 4) ? 8 : 0);
    const __bf16* p = tile + r * 128 + ka;
    v8bf lo = *(const v8bf*)p;
    v8bf hi = *(const v8bf*)(p + 16);
    return bf_combine(lo, hi);
}

// B fragment: 32x16 bf16 from transposed weight wT[n][k] (stride elems per n-row).
// lanes 0-15: N=lane, K=k0..k0+7 , k0+16..k0+23 ; lanes 16-31: N=lane-16, K=k0+8..k0+15 , k0+24..k0+31
__device__ inline v16bf load_b_frag(const __bf16* wT, int stride, int nb, int k0, int lane) {
    int n = nb + (lane & 15);
    int k = k0 + ((lane >> 4) ? 8 : 0);
    const __bf16* p = wT + n * stride + k;
    return bf_combine(*(const v8bf*)p, *(const v8bf*)(p + 16));
}

// ---------------------------------------------------------------- TDM (Tensor Data Mover)
// D# per CDNA5 ISA. data_size = 4B. 2D tile (tile_dim1==0 -> 1D).
__device__ inline void tdm_load_to_lds(unsigned long long gaddr, unsigned int laddr,
                                       unsigned int tdim0, unsigned int tdim1,
                                       unsigned int tiled0, unsigned int tiled1,
                                       unsigned long long stride0) {
    v4u g0;
    g0[0] = 1u;                                            // count=1, user mode
    g0[1] = laddr;                                         // lds_addr
    g0[2] = (unsigned int)(gaddr & 0xFFFFFFFFull);         // global_addr[31:0]
    g0[3] = (unsigned int)((gaddr >> 32) & 0x1FFFFFFull)   // global_addr[56:32]
          | (2u << 30);                                    // type = 2 ("image")
    v8i g1;
    g1[0] = (int)(2u << 16);                               // wg_mask=0, data_size=2 (4B)
    g1[1] = (int)((tdim0 & 0xFFFFu) << 16);                // [63:48] tensor_dim0 lo
    g1[2] = (int)(((tdim0 >> 16) & 0xFFFFu) | ((tdim1 & 0xFFFFu) << 16));
    g1[3] = (int)(((tdim1 >> 16) & 0xFFFFu) | ((tiled0 & 0xFFFFu) << 16)); // tile_dim0
    g1[4] = (int)(tiled1 & 0xFFFFu);                       // tile_dim1 (tile_dim2 = 0)
    g1[5] = (int)(stride0 & 0xFFFFFFFFull);                // tensor_dim0_stride lo
    g1[6] = (int)((stride0 >> 32) & 0xFFFFull);            // stride hi; dim1_stride=0
    g1[7] = 0;
    v4i z4 = {0, 0, 0, 0};
#if defined(__clang_major__) && (__clang_major__ >= 23)
    v8i z8 = {0, 0, 0, 0, 0, 0, 0, 0};
    __builtin_amdgcn_tensor_load_to_lds(g0, g1, z4, z4, z8, 0);
#else
    __builtin_amdgcn_tensor_load_to_lds(g0, g1, z4, z4, 0);
#endif
}

// ---------------------------------------------------------------- GCN part
__global__ void k_init_deg(float* deg) {
    int i = blockIdx.x * blockDim.x + threadIdx.x;
    if (i < N_NODES) deg[i] = 1.0f;   // self-loop
}

__global__ void k_deg_edges(const int* dst, float* deg) {
    int e = blockIdx.x * blockDim.x + threadIdx.x;
    if (e < E_EDGES) atomicAdd(&deg[dst[e]], 1.0f);
}

__global__ void k_dinv(const float* deg, float* dinv) {
    int i = blockIdx.x * blockDim.x + threadIdx.x;
    if (i < N_NODES) dinv[i] = rsqrtf(deg[i]);
}

__global__ void k_gemm_nn(const float* __restrict__ A, const float* __restrict__ W,
                          float* __restrict__ C, int n, int din, int dout) {
    int idx = blockIdx.x * blockDim.x + threadIdx.x;
    if (idx >= n * dout) return;
    int r = idx / dout, c = idx % dout;
    const float* a = A + r * din;
    float s = 0.f;
    for (int k = 0; k < din; ++k) s += a[k] * W[k * dout + c];
    C[idx] = s;
}

__global__ void k_init_agg(const float* m, const float* dinv, const float* b,
                           float* agg, int n, int d) {
    int idx = blockIdx.x * blockDim.x + threadIdx.x;
    if (idx >= n * d) return;
    int r = idx / d, c = idx % d;
    float w = dinv[r] * dinv[r];
    agg[idx] = b[c] + w * m[idx];
}

__global__ void k_scatter(const float* __restrict__ m, const int* __restrict__ src,
                          const int* __restrict__ dst, const float* __restrict__ dinv,
                          float* agg, int d) {
    long idx = (long)blockIdx.x * blockDim.x + threadIdx.x;
    if (idx >= (long)E_EDGES * d) return;
    int e = (int)(idx / d), c = (int)(idx % d);
    int s = src[e], t = dst[e];
    atomicAdd(&agg[t * d + c], m[s * d + c] * dinv[s] * dinv[t]);
}

__global__ void k_relu(float* x, int n) {
    int i = blockIdx.x * blockDim.x + threadIdx.x;
    if (i < n) x[i] = fmaxf(x[i], 0.f);
}

// ------------------------------------------------------- attention (closed form)
__global__ void k_attn_pre(const float* Wq, const float* bq, const float* Wk, const float* bk,
                           const float* Wv, const float* bv, const float* Wo, const float* bo,
                           float* ABCD, float* M, float* c0) {
    __shared__ float cq[HID], ck[HID], cv[HID];
    int j = threadIdx.x;
    float sq = 0.f, sk = 0.f, sv = 0.f;
    for (int i = 0; i < HID; ++i) {
        sq += Wq[i * HID + j]; sk += Wk[i * HID + j]; sv += Wv[i * HID + j];
    }
    cq[j] = sq; ck[j] = sk; cv[j] = sv;
    __syncthreads();
    if (j < HEADS) {
        float A = 0.f, B = 0.f, C = 0.f, D = 0.f;
        for (int d = 0; d < DH; ++d) {
            int t = j * DH + d;
            A += cq[t] * ck[t]; B += cq[t] * bk[t];
            C += ck[t] * bq[t]; D += bq[t] * bk[t];
        }
        ABCD[0 * HEADS + j] = A; ABCD[1 * HEADS + j] = B;
        ABCD[2 * HEADS + j] = C; ABCD[3 * HEADS + j] = D;
    }
    for (int h = 0; h < HEADS; ++h) {
        float s = 0.f;
        for (int d = 0; d < DH; ++d) { int t = h * DH + d; s += cv[t] * Wo[t * HID + j]; }
        M[h * HID + j] = s;
    }
    float s0 = 0.f;
    for (int i = 0; i < HID; ++i) s0 += bv[i] * Wo[i * HID + j];
    c0[j] = s0 + bo[j];
}

__global__ void k_attn_node(const float* __restrict__ h3, const float* __restrict__ ABCD,
                            float* __restrict__ S) {
    __shared__ float hn[OUT_CH];
    int n = blockIdx.x, t = threadIdx.x;
    if (t < OUT_CH) hn[t] = h3[n * OUT_CH + t];
    __syncthreads();
    int head = t >> 5, q = t & 31;
    float A = ABCD[head], B = ABCD[8 + head], C = ABCD[16 + head], D = ABCD[24 + head];
    float hq = hn[q];
    float e[OUT_CH];
    float mx = -1e30f;
#pragma unroll
    for (int k = 0; k < OUT_CH; ++k) {
        float sc = (A * hq * hn[k] + B * hq + C * hn[k] + D) * 0.25f; // 1/sqrt(16)
        e[k] = sc; mx = fmaxf(mx, sc);
    }
    float sum = 0.f, sv = 0.f;
#pragma unroll
    for (int k = 0; k < OUT_CH; ++k) {
        float w = expf(e[k] - mx);
        sum += w; sv += w * hn[k];
    }
    S[(n * OUT_CH + q) * HEADS + head] = sv / sum;
}

__global__ void k_x1_ln(const float* __restrict__ h3, const float* __restrict__ S,
                        const float* __restrict__ M, const float* __restrict__ c0,
                        const float* __restrict__ g, const float* __restrict__ be,
                        float* __restrict__ x1) {
    int row = blockIdx.x;               // n*32+q
    int j = threadIdx.x;
    float a = c0[j];
    const float* Sp = S + row * HEADS;
#pragma unroll
    for (int h = 0; h < HEADS; ++h) a += Sp[h] * M[h * HID + j];
    float v = h3[row] + a;
    __shared__ float red[HID];
    red[j] = v; __syncthreads();
    for (int s = 64; s > 0; s >>= 1) { if (j < s) red[j] += red[j + s]; __syncthreads(); }
    float mu = red[0] * (1.f / HID);
    __syncthreads();
    float d = v - mu;
    red[j] = d * d; __syncthreads();
    for (int s = 64; s > 0; s >>= 1) { if (j < s) red[j] += red[j + s]; __syncthreads(); }
    float var = red[0] * (1.f / HID);
    x1[row * HID + j] = d * rsqrtf(var + EPS) * g[j] + be[j];
}

// ------------------------------------------------------- weight transposes (bf16)
__global__ void k_prep_w1t(const float* __restrict__ W1, __bf16* __restrict__ W1T) {
    int idx = blockIdx.x * blockDim.x + threadIdx.x;   // 128*2048
    if (idx >= HID * FF) return;
    int i = idx / FF, jc = idx % FF;
    W1T[jc * HID + i] = (__bf16)W1[idx];
}
__global__ void k_prep_w2t(const float* __restrict__ W2, __bf16* __restrict__ W2T) {
    int idx = blockIdx.x * blockDim.x + threadIdx.x;   // 2048*128
    if (idx >= FF * HID) return;
    int i = idx / HID, j = idx % HID;
    W2T[j * FF + i] = (__bf16)W2[idx];
}

// ------------------------------------------------------- fused FFN + residual + LN2
// 128 rows per workgroup, 8 waves; wave w owns rows w*16..w*16+15, all 128 output cols.
// FF processed in 16 chunks of 128. Weight chunks staged in LDS via TDM, pipelined:
//   W1c(c+1) TDM overlaps GEMM2(c); W2c(c+1) TDM issued after GEMM2(c) completes.
// A fragments of Xs are chunk-invariant -> loaded ONCE per kernel (32 VGPRs).
// GEMM2 runs kt-outer so its 8 WMMAs per kt are independent (no C-chains).
// LDS map (dynamic, 128KB): [0,32K) Xs bf16 | [32K,64K) Hs bf16 |
//                           [64K,96K) W1c bf16 | [96K,128K) W2c bf16
//                           (Os f32 overlays [64K,128K) after the last chunk)
__global__ void __launch_bounds__(256)
k_ffn(const float* __restrict__ x1, const __bf16* __restrict__ w1t,
      const __bf16* __restrict__ w2t, const float* __restrict__ b1,
      const float* __restrict__ b2, const float* __restrict__ g2,
      const float* __restrict__ be2, float* __restrict__ out) {
    extern __shared__ __align__(16) char smem[];
    __bf16* Xs  = (__bf16*)smem;                // 128x128 bf16
    __bf16* Hs  = (__bf16*)(smem + 32768);      // 128x128 bf16
    __bf16* W1c = (__bf16*)(smem + 65536);      // 128x128 bf16 (chunk of W1T)
    __bf16* W2c = (__bf16*)(smem + 98304);      // 128x128 bf16 (chunk of W2T window)
    float*  Os  = (float*)(smem + 65536);       // 128x128 f32, overlays W1c/W2c

    int tid  = threadIdx.x;
    int lane = tid & 31;
    int wave = tid >> 5;        // 0..7 -> row stripe
    int rowBase = blockIdx.x * 128;

    unsigned int l_w1 = (unsigned int)(unsigned long long)(uintptr_t)(void*)W1c;
    unsigned int l_w2 = (unsigned int)(unsigned long long)(uintptr_t)(void*)W2c;

    // stage X tile as bf16 (vectorized: float4 -> 4x bf16)
    {
        const float4* xv = (const float4*)(x1 + (size_t)rowBase * 128);
        v4bf* xs4 = (v4bf*)Xs;
        for (int i = tid; i < (128 * 128) / 4; i += 256) {
            float4 f = xv[i];
            v4bf o;
            o[0] = (__bf16)f.x; o[1] = (__bf16)f.y;
            o[2] = (__bf16)f.z; o[3] = (__bf16)f.w;
            xs4[i] = o;
        }
    }

    // prime chunk 0 weights with TDM (wave 0 issues; TDM ignores EXEC)
    if (wave == 0) {
        // W1 chunk: contiguous 32KB = 8192 x 4B, 1D
        tdm_load_to_lds((unsigned long long)(uintptr_t)(w1t), l_w1,
                        8192u, 1u, 8192u, 0u, 8192ull);
        // W2 chunk: 128 rows x 64 x 4B, row stride 1024 x 4B
        tdm_load_to_lds((unsigned long long)(uintptr_t)(w2t), l_w2,
                        64u, 128u, 64u, 128u, 1024ull);
        __builtin_amdgcn_s_wait_tensorcnt(0);
    }
    __syncthreads();

    // Xs A-fragments: invariant across all chunks -> load once
    v16bf ax[4];
#pragma unroll
    for (int kt = 0; kt < 4; ++kt) ax[kt] = load_a_frag(Xs, wave * 16, kt * 32, lane);

    v8f acc[8];
#pragma unroll
    for (int t = 0; t < 8; ++t) acc[t] = (v8f){0.f,0.f,0.f,0.f,0.f,0.f,0.f,0.f};

    for (int chunk = 0; chunk < FF / 128; ++chunk) {
        // ---- GEMM1: 16x128 stripe per wave, K=128; bias+relu -> Hs
#pragma unroll
        for (int nt = 0; nt < 8; ++nt) {
            int nb = nt * 16;
            v8f c = (v8f){0.f,0.f,0.f,0.f,0.f,0.f,0.f,0.f};
#pragma unroll
            for (int kt = 0; kt < 4; ++kt) {
                v16bf b = load_b_frag(W1c, 128, nb, kt * 32, lane);
                c = __builtin_amdgcn_wmma_f32_16x16x32_bf16(false, ax[kt], false, b,
                                                            (short)0, c, false, false);
            }
            int col = nb + (lane & 15);
            int rb  = wave * 16 + ((lane >> 4) ? 8 : 0);
            float bb = b1[chunk * 128 + col];
#pragma unroll
            for (int i = 0; i < 8; ++i)
                Hs[(rb + i) * 128 + col] = (__bf16)fmaxf(c[i] + bb, 0.f);
        }
        __syncthreads();   // Hs ready; all waves done reading W1c

        // prefetch next W1 chunk via TDM, overlapping GEMM2
        if (wave == 0 && chunk + 1 < FF / 128) {
            tdm_load_to_lds((unsigned long long)(uintptr_t)(w1t + (chunk + 1) * 128 * 128),
                            l_w1, 8192u, 1u, 8192u, 0u, 8192ull);
        }

        // ---- GEMM2: acc += Hs @ W2c ; kt-outer => 8 independent WMMAs per step
        {
            v16bf ah[4];
#pragma unroll
            for (int kt = 0; kt < 4; ++kt) ah[kt] = load_a_frag(Hs, wave * 16, kt * 32, lane);
#pragma unroll
            for (int kt = 0; kt < 4; ++kt) {
#pragma unroll
                for (int nt = 0; nt < 8; ++nt) {
                    v16bf b = load_b_frag(W2c, 128, nt * 16, kt * 32, lane);
                    acc[nt] = __builtin_amdgcn_wmma_f32_16x16x32_bf16(false, ah[kt], false, b,
                                                                      (short)0, acc[nt], false, false);
                }
            }
        }
        __syncthreads();   // all waves done reading W2c / Hs

        if (wave == 0 && chunk + 1 < FF / 128) {
            tdm_load_to_lds((unsigned long long)(uintptr_t)(w2t + (chunk + 1) * 128),
                            l_w2, 64u, 128u, 64u, 128u, 1024ull);
            __builtin_amdgcn_s_wait_tensorcnt(0);   // both pending chunks complete
        }
        __syncthreads();
    }

    // stage accumulators into Os (overlays weight buffers; safe after final barrier)
#pragma unroll
    for (int nt = 0; nt < 8; ++nt) {
        int col = nt * 16 + (lane & 15);
        int rb  = wave * 16 + ((lane >> 4) ? 8 : 0);
#pragma unroll
        for (int i = 0; i < 8; ++i)
            Os[(rb + i) * 128 + col] = acc[nt][i];
    }
    __syncthreads();

    // residual + LN2, one thread per row
    if (tid < 128) {
        int row = rowBase + tid;
        const float* xr = x1 + (size_t)row * 128;
        const float* orow = Os + tid * 128;
        float s = 0.f, ss = 0.f;
        for (int j = 0; j < 128; ++j) {
            float v = xr[j] + orow[j] + b2[j];
            s += v; ss += v * v;
        }
        float mu  = s * (1.f / 128.f);
        float var = ss * (1.f / 128.f) - mu * mu;
        float inv = rsqrtf(var + EPS);
        for (int j = 0; j < 128; ++j) {
            float v = xr[j] + orow[j] + b2[j];
            out[(size_t)row * 128 + j] = (v - mu) * inv * g2[j] + be2[j];
        }
    }
}

// ---------------------------------------------------------------- launcher
extern "C" void kernel_launch(void* const* d_in, const int* in_sizes, int n_in,
                              void* d_out, int out_size, void* d_ws, size_t ws_size,
                              hipStream_t stream) {
    const float* x   = (const float*)d_in[0];
    const int*   ei  = (const int*)d_in[1];
    const int*   src = ei;
    const int*   dst = ei + E_EDGES;
    const float* Wg1 = (const float*)d_in[2];  const float* bg1 = (const float*)d_in[3];
    const float* Wg2 = (const float*)d_in[4];  const float* bg2 = (const float*)d_in[5];
    const float* Wg3 = (const float*)d_in[6];  const float* bg3 = (const float*)d_in[7];
    const float* Wq  = (const float*)d_in[8];  const float* bq  = (const float*)d_in[9];
    const float* Wk  = (const float*)d_in[10]; const float* bk  = (const float*)d_in[11];
    const float* Wv  = (const float*)d_in[12]; const float* bv  = (const float*)d_in[13];
    const float* Wo  = (const float*)d_in[14]; const float* bo  = (const float*)d_in[15];
    const float* l1g = (const float*)d_in[16]; const float* l1b = (const float*)d_in[17];
    const float* W1  = (const float*)d_in[18]; const float* b1  = (const float*)d_in[19];
    const float* W2  = (const float*)d_in[20]; const float* b2  = (const float*)d_in[21];
    const float* l2g = (const float*)d_in[22]; const float* l2b = (const float*)d_in[23];
    float* out = (float*)d_out;

    char* p = (char*)d_ws;
    auto alloc = [&](size_t bytes) -> char* {
        char* r = p; p += (bytes + 255) & ~(size_t)255; return r;
    };
    float* deg  = (float*)alloc(N_NODES * 4);
    float* dinv = (float*)alloc(N_NODES * 4);
    float* mbuf = (float*)alloc((size_t)N_NODES * HID * 4);
    float* h1   = (float*)alloc((size_t)N_NODES * HID * 4);
    float* h2   = (float*)alloc((size_t)N_NODES * HID * 4);
    float* h3   = (float*)alloc((size_t)N_NODES * OUT_CH * 4);
    float* ABCD = (float*)alloc(4 * HEADS * 4);
    float* Mh   = (float*)alloc(HEADS * HID * 4);
    float* c0   = (float*)alloc(HID * 4);
    float* S    = (float*)alloc((size_t)N_NODES * OUT_CH * HEADS * 4);
    float* x1   = (float*)alloc((size_t)N_NODES * OUT_CH * HID * 4);
    __bf16* w1t = (__bf16*)alloc((size_t)FF * HID * 2);
    __bf16* w2t = (__bf16*)alloc((size_t)FF * HID * 2);

    k_prep_w1t<<<(HID * FF + 255) / 256, 256, 0, stream>>>(W1, w1t);
    k_prep_w2t<<<(FF * HID + 255) / 256, 256, 0, stream>>>(W2, w2t);

    k_init_deg<<<(N_NODES + 255) / 256, 256, 0, stream>>>(deg);
    k_deg_edges<<<(E_EDGES + 255) / 256, 256, 0, stream>>>(dst, deg);
    k_dinv<<<(N_NODES + 255) / 256, 256, 0, stream>>>(deg, dinv);

    // GCN layer 1: 128 -> 128
    k_gemm_nn<<<(N_NODES * HID + 255) / 256, 256, 0, stream>>>(x, Wg1, mbuf, N_NODES, C_IN, HID);
    k_init_agg<<<(N_NODES * HID + 255) / 256, 256, 0, stream>>>(mbuf, dinv, bg1, h1, N_NODES, HID);
    k_scatter<<<(int)(((long)E_EDGES * HID + 255) / 256), 256, 0, stream>>>(mbuf, src, dst, dinv, h1, HID);
    k_relu<<<(N_NODES * HID + 255) / 256, 256, 0, stream>>>(h1, N_NODES * HID);
    // GCN layer 2: 128 -> 128
    k_gemm_nn<<<(N_NODES * HID + 255) / 256, 256, 0, stream>>>(h1, Wg2, mbuf, N_NODES, HID, HID);
    k_init_agg<<<(N_NODES * HID + 255) / 256, 256, 0, stream>>>(mbuf, dinv, bg2, h2, N_NODES, HID);
    k_scatter<<<(int)(((long)E_EDGES * HID + 255) / 256), 256, 0, stream>>>(mbuf, src, dst, dinv, h2, HID);
    k_relu<<<(N_NODES * HID + 255) / 256, 256, 0, stream>>>(h2, N_NODES * HID);
    // GCN layer 3: 128 -> 32
    k_gemm_nn<<<(N_NODES * OUT_CH + 255) / 256, 256, 0, stream>>>(h2, Wg3, mbuf, N_NODES, HID, OUT_CH);
    k_init_agg<<<(N_NODES * OUT_CH + 255) / 256, 256, 0, stream>>>(mbuf, dinv, bg3, h3, N_NODES, OUT_CH);
    k_scatter<<<(int)(((long)E_EDGES * OUT_CH + 255) / 256), 256, 0, stream>>>(mbuf, src, dst, dinv, h3, OUT_CH);
    k_relu<<<(N_NODES * OUT_CH + 255) / 256, 256, 0, stream>>>(h3, N_NODES * OUT_CH);

    // attention (closed form thanks to rank-1 broadcast of t)
    k_attn_pre<<<1, 128, 0, stream>>>(Wq, bq, Wk, bk, Wv, bv, Wo, bo, ABCD, Mh, c0);
    k_attn_node<<<N_NODES, 256, 0, stream>>>(h3, ABCD, S);
    k_x1_ln<<<N_NODES * OUT_CH, 128, 0, stream>>>(h3, S, Mh, c0, l1g, l1b, x1);

    // fused FFN + residual + LN2 (WMMA bf16, TDM-staged weights, 128KB dynamic LDS)
    k_ffn<<<(N_NODES * OUT_CH) / 128, 256, 131072, stream>>>(x1, w1t, w2t, b1, b2, l2g, l2b, out);
}
